// GraphNet_4260607557736
// MI455X (gfx1250) — matrix operand order repaired and checked by reading, not compiled, not run on hardware
//
#include <hip/hip_runtime.h>
#include <hip/hip_bf16.h>

// CDNA5 wave32 WMMA fragment types
typedef __attribute__((ext_vector_type(16))) __bf16 v16bf;
typedef __attribute__((ext_vector_type(8)))  __bf16 v8bf;
typedef __attribute__((ext_vector_type(8)))  float  v8f;

#define N_NODES   65536
#define NEDGES    524288
#define BATCH     64
#define NODES_PER 1024   // N_NODES / BATCH

// ---------------------------------------------------------------------------
// z = h (strided source copy), float4 vectorized (C always multiple of 4)
// ---------------------------------------------------------------------------
__global__ void k_copy(const float* __restrict__ src, int lds_,
                       float* __restrict__ dst, int C, int M) {
  int gid = blockIdx.x * blockDim.x + threadIdx.x;
  int cpr = C >> 2;
  int total = M * cpr;
  if (gid >= total) return;
  int r = gid / cpr, c4 = (gid - r * cpr) << 2;
  *(float4*)&dst[(long)r * C + c4] =
      *(const float4*)&src[(long)r * lds_ + c4];
}

// ---------------------------------------------------------------------------
// z[dst] += h[src]  (float segment-sum via global atomics; 4 channels/thread)
// ---------------------------------------------------------------------------
__global__ void k_edge(const int* __restrict__ es, const int* __restrict__ ed,
                       const float* __restrict__ h, int ldh,
                       float* __restrict__ z, int C) {
  long gid = (long)blockIdx.x * blockDim.x + threadIdx.x;
  int groups = C >> 2;
  long total = (long)NEDGES * groups;
  if (gid >= total) return;
  int e = (int)(gid / groups);
  int g = (int)(gid - (long)e * groups) << 2;
  int s = es[e], d = ed[e];
  float4 hv = *(const float4*)(h + (long)s * ldh + g);
  float* zp = z + (long)d * C + g;
  atomicAdd(zp + 0, hv.x);
  atomicAdd(zp + 1, hv.y);
  atomicAdd(zp + 2, hv.z);
  atomicAdd(zp + 3, hv.w);
}

// ---------------------------------------------------------------------------
// Generic WMMA GEMM:  C[M x N] = epilogue( A[M x K] @ W[K x N] (+ C if accum) )
//   epilogue: +bias, eval-BatchNorm(g,b,m,v), ReLU  (any subset)
// Block = 256 threads = 8 waves; block tile 64(M) x 64(N); K chunk = 32.
// Each wave owns a 16x32 C tile -> 2 v_wmma_f32_16x16x32_bf16 per chunk.
// A staged row-major bf16 in LDS; B staged TRANSPOSED (Bt[n][k]) so both
// fragment reads are contiguous 16B ds_load_b128s. All guards branchless
// (K is always a multiple of 16; staging groups are 8-aligned).
// ---------------------------------------------------------------------------
__global__ __launch_bounds__(256)
void k_gemm(const float* __restrict__ A, int lda,
            const float* __restrict__ W,
            float* __restrict__ Cp, int ldc,
            int K, int N,
            const float* __restrict__ bias,
            const float* __restrict__ bng, const float* __restrict__ bnb,
            const float* __restrict__ bnm, const float* __restrict__ bnv,
            int relu, int accum) {
  __shared__ __bf16 As[64][40];   // row stride 80B (16B aligned)
  __shared__ __bf16 Bt[64][40];   // TRANSPOSED: Bt[n][k]

  const int tid  = threadIdx.x;
  const int lane = tid & 31;
  const int wid  = tid >> 5;     // 0..7
  const int wm   = wid & 3;      // M sub-tile (0..3)
  const int wn   = wid >> 2;     // N sub-tile (0..1), 32 cols each
  const int hl   = lane >> 4;    // lane half (0/1)
  const int l16  = lane & 15;

  const int rowBase = blockIdx.x * 64;
  const int colBase = blockIdx.y * 64;

  // A staging: thread -> (row ar, 8 cols from ac); 64x32 chunk
  const int ar = tid >> 2;
  const int ac = (tid & 3) << 3;
  // B staging: thread -> (col bn_, 8 ks from bk); stored transposed
  const int bn_ = tid & 63;
  const int bk  = (tid >> 6) << 3;
  const int bcol  = colBase + bn_;
  const bool bnok = bcol < N;
  const int bcolC = bnok ? bcol : 0;

  const int grow0 = rowBase + wm * 16 + hl * 8;  // first of 8 rows per lane
  const int gcol0 = colBase + wn * 32 + l16;     // frag 0 column
  const int gcol1 = gcol0 + 16;                  // frag 1 column

  v8f acc0, acc1;
#pragma unroll
  for (int r = 0; r < 8; ++r) { acc0[r] = 0.0f; acc1[r] = 0.0f; }
  if (accum) {
    if (gcol0 < N) {
#pragma unroll
      for (int r = 0; r < 8; ++r) acc0[r] = Cp[(long)(grow0 + r) * ldc + gcol0];
    }
    if (gcol1 < N) {
#pragma unroll
      for (int r = 0; r < 8; ++r) acc1[r] = Cp[(long)(grow0 + r) * ldc + gcol1];
    }
  }

  for (int k0 = 0; k0 < K; k0 += 32) {
    // ---- stage A: 2x float4 loads -> 8 bf16 -> one ds_store_b128 ----
    {
      const bool ok = (k0 + ac) < K;  // 8-group wholly valid or not (K%16==0)
      const float* ap = A + (long)(rowBase + ar) * lda + (ok ? (k0 + ac) : 0);
      float4 u = *(const float4*)(ap);
      float4 w4 = *(const float4*)(ap + 4);
      v8bf pk;
      pk[0] = (__bf16)(ok ? u.x  : 0.f); pk[1] = (__bf16)(ok ? u.y  : 0.f);
      pk[2] = (__bf16)(ok ? u.z  : 0.f); pk[3] = (__bf16)(ok ? u.w  : 0.f);
      pk[4] = (__bf16)(ok ? w4.x : 0.f); pk[5] = (__bf16)(ok ? w4.y : 0.f);
      pk[6] = (__bf16)(ok ? w4.z : 0.f); pk[7] = (__bf16)(ok ? w4.w : 0.f);
      *(v8bf*)&As[ar][ac] = pk;
    }
    // ---- stage B transposed: 8 coalesced b32 loads -> one ds_store_b128 ----
    {
      const bool kok = (k0 + bk) < K;
      const int kb = kok ? (k0 + bk) : 0;
      const bool ok = kok && bnok;
      v8bf pk;
#pragma unroll
      for (int j = 0; j < 8; ++j) {
        float v = W[(long)(kb + j) * N + bcolC];
        pk[j] = (__bf16)(ok ? v : 0.f);
      }
      *(v8bf*)&Bt[bn_][bk] = pk;
    }
    __syncthreads();

    // ---- fragments: contiguous 16B LDS vector loads ----
    const int arow = wm * 16 + l16;
    v8bf alo = *(const v8bf*)&As[arow][hl * 8];
    v8bf ahi = *(const v8bf*)&As[arow][16 + hl * 8];
    v16bf af = __builtin_shufflevector(alo, ahi, 0, 1, 2, 3, 4, 5, 6, 7,
                                       8, 9, 10, 11, 12, 13, 14, 15);
    {
      const int bc = wn * 32 + l16;
      v8bf blo = *(const v8bf*)&Bt[bc][hl * 8];
      v8bf bhi = *(const v8bf*)&Bt[bc][16 + hl * 8];
      v16bf bf0 = __builtin_shufflevector(blo, bhi, 0, 1, 2, 3, 4, 5, 6, 7,
                                          8, 9, 10, 11, 12, 13, 14, 15);
      acc0 = __builtin_amdgcn_wmma_f32_16x16x32_bf16(
          false, af, false, bf0, (short)0, acc0, false, false);
    }
    {
      const int bc = wn * 32 + 16 + l16;
      v8bf blo = *(const v8bf*)&Bt[bc][hl * 8];
      v8bf bhi = *(const v8bf*)&Bt[bc][16 + hl * 8];
      v16bf bf1 = __builtin_shufflevector(blo, bhi, 0, 1, 2, 3, 4, 5, 6, 7,
                                          8, 9, 10, 11, 12, 13, 14, 15);
      acc1 = __builtin_amdgcn_wmma_f32_16x16x32_bf16(
          false, af, false, bf1, (short)0, acc1, false, false);
    }
    __syncthreads();
  }

  // ---- epilogue ----
#pragma unroll
  for (int f = 0; f < 2; ++f) {
    int gcol = f ? gcol1 : gcol0;
    v8f acc = f ? acc1 : acc0;
    if (gcol < N) {
      float bia = bias ? bias[gcol] : 0.0f;
      float g = 1.f, bb = 0.f, mm = 0.f, inv = 1.f;
      if (bng) {
        g = bng[gcol]; bb = bnb[gcol]; mm = bnm[gcol];
        inv = rsqrtf(bnv[gcol] + 1e-5f);
      }
#pragma unroll
      for (int r = 0; r < 8; ++r) {
        float v = acc[r] + bia;
        if (bng) v = g * (v - mm) * inv + bb;
        if (relu) v = v > 0.f ? v : 0.f;
        Cp[(long)(grow0 + r) * ldc + gcol] = v;
      }
    }
  }
}

// ---------------------------------------------------------------------------
// PyG GraphNorm, batch stats over dim 0 of [BATCH, NODES_PER, C], in place
// ---------------------------------------------------------------------------
__global__ void k_graphnorm(float* __restrict__ a, const float* __restrict__ g,
                            const float* __restrict__ b,
                            const float* __restrict__ alpha, int C) {
  int idx = blockIdx.x * blockDim.x + threadIdx.x;
  int total = NODES_PER * C;
  if (idx >= total) return;
  int j = idx / C, c = idx - j * C;
  long stride = (long)NODES_PER * C;
  float* p = a + (long)j * C + c;
  float s = 0.f;
  for (int bb = 0; bb < BATCH; ++bb) s += p[bb * stride];
  float mean = s * (1.0f / BATCH);
  float al = alpha[c];
  float m2 = 0.f;
  for (int bb = 0; bb < BATCH; ++bb) {
    float o = p[bb * stride] - al * mean;
    m2 += o * o;
  }
  float inv = rsqrtf(m2 * (1.0f / BATCH) + 1e-5f);
  float gg = g[c], bv = b[c];
  for (int bb = 0; bb < BATCH; ++bb) {
    float o = p[bb * stride] - al * mean;
    p[bb * stride] = gg * o * inv + bv;
  }
}

// ---------------------------------------------------------------------------
// Value head: mean over nodes ([B,1024,3] -> [B,3]) then log_softmax(dim=1)
// ---------------------------------------------------------------------------
__global__ void k_vhead(const float* __restrict__ vout, float* __restrict__ out) {
  __shared__ float red[3][128];
  int b = blockIdx.x, t = threadIdx.x;
  float s0 = 0.f, s1 = 0.f, s2 = 0.f;
  for (int j = t; j < NODES_PER; j += 128) {
    const float* p = vout + ((long)b * NODES_PER + j) * 3;
    s0 += p[0]; s1 += p[1]; s2 += p[2];
  }
  red[0][t] = s0; red[1][t] = s1; red[2][t] = s2;
  __syncthreads();
  for (int off = 64; off > 0; off >>= 1) {
    if (t < off) {
      red[0][t] += red[0][t + off];
      red[1][t] += red[1][t + off];
      red[2][t] += red[2][t + off];
    }
    __syncthreads();
  }
  if (t == 0) {
    float v0 = red[0][0] * (1.0f / NODES_PER);
    float v1 = red[1][0] * (1.0f / NODES_PER);
    float v2 = red[2][0] * (1.0f / NODES_PER);
    float mx = fmaxf(v0, fmaxf(v1, v2));
    float lse = logf(expf(v0 - mx) + expf(v1 - mx) + expf(v2 - mx));
    out[b * 3 + 0] = v0 - mx - lse;
    out[b * 3 + 1] = v1 - mx - lse;
    out[b * 3 + 2] = v2 - mx - lse;
  }
}

// ---------------------------------------------------------------------------
// Policy head: log_softmax over 1024 nodes per batch row
// ---------------------------------------------------------------------------
__global__ void k_phead(const float* __restrict__ pout, float* __restrict__ out) {
  __shared__ float red[256];
  int b = blockIdx.x, t = threadIdx.x;
  const float* p = pout + (long)b * NODES_PER;
  float mx = -3.4e38f;
  for (int j = t; j < NODES_PER; j += 256) mx = fmaxf(mx, p[j]);
  red[t] = mx; __syncthreads();
  for (int off = 128; off > 0; off >>= 1) {
    if (t < off) red[t] = fmaxf(red[t], red[t + off]);
    __syncthreads();
  }
  mx = red[0]; __syncthreads();
  float s = 0.f;
  for (int j = t; j < NODES_PER; j += 256) s += expf(p[j] - mx);
  red[t] = s; __syncthreads();
  for (int off = 128; off > 0; off >>= 1) {
    if (t < off) red[t] += red[t + off];
    __syncthreads();
  }
  float lse = logf(red[0]);
  for (int j = t; j < NODES_PER; j += 256)
    out[(long)b * NODES_PER + j] = p[j] - mx - lse;
}

// ---------------------------------------------------------------------------
extern "C" void kernel_launch(void* const* d_in, const int* in_sizes, int n_in,
                              void* d_out, int out_size, void* d_ws, size_t ws_size,
                              hipStream_t stream) {
  // --- input mapping: setup_inputs() dict-insertion order, nested dicts too ---
  int ii = 0;
  const float* X = (const float*)d_in[ii++];                       // x [65536,16]
  const float *cw1[4], *cb1[4], *cg[4], *cbb[4], *cm[4], *cv[4], *cw2[4], *cb2[4];
  for (int l = 0; l < 4; ++l) {                                    // convs: w1,b1,bn{g,b,m,v},w2,b2
    cw1[l] = (const float*)d_in[ii++];  cb1[l] = (const float*)d_in[ii++];
    cg[l]  = (const float*)d_in[ii++];  cbb[l] = (const float*)d_in[ii++];
    cm[l]  = (const float*)d_in[ii++];  cv[l]  = (const float*)d_in[ii++];
    cw2[l] = (const float*)d_in[ii++];  cb2[l] = (const float*)d_in[ii++];
  }
  const float *ng[4], *nb[4], *nm[4], *nv[4];
  for (int l = 0; l < 4; ++l) {                                    // norms: g,b,m,v
    ng[l] = (const float*)d_in[ii++];  nb[l] = (const float*)d_in[ii++];
    nm[l] = (const float*)d_in[ii++];  nv[l] = (const float*)d_in[ii++];
  }
  const float* jkw = (const float*)d_in[ii++];   // [512,512]
  const float* jkb = (const float*)d_in[ii++];
  const float* l1w = (const float*)d_in[ii++];   // [528,256]
  const float* l1b = (const float*)d_in[ii++];
  const float* gng = (const float*)d_in[ii++];   // gn: g,b,alpha
  const float* gnb = (const float*)d_in[ii++];
  const float* gna = (const float*)d_in[ii++];
  const float* l2w = (const float*)d_in[ii++];   // [256,128]
  const float* l2b = (const float*)d_in[ii++];
  const float* v1w = (const float*)d_in[ii++];
  const float* v1b = (const float*)d_in[ii++];
  const float* v2w = (const float*)d_in[ii++];
  const float* v2b = (const float*)d_in[ii++];
  const float* p1w = (const float*)d_in[ii++];
  const float* p1b = (const float*)d_in[ii++];
  const float* p2w = (const float*)d_in[ii++];
  const float* p2b = (const float*)d_in[ii++];
  const int* edge  = (const int*)d_in[ii++];     // [2, 524288]
  const int* esrc = edge;
  const int* edst = edge + NEDGES;

  // --- workspace layout (floats), with buffer reuse ---
  float* ws   = (float*)d_ws;
  float* xs   = ws;                                   // [65536,512] JK concat
  float* jk   = xs   + (long)N_NODES * 512;           // [65536,512]
  float* bufA = jk   + (long)N_NODES * 512;           // [65536,256]: z/t -> a1 -> heads
  float* bufB = bufA + (long)N_NODES * 256;           // [65536,128]: h2
  float* z    = bufA;                                 // z (<=128 ch)
  float* t    = bufA + (long)N_NODES * 128;           // MLP hidden
  float* a1   = bufA;                                 // [65536,256] after l1
  float* vhid = bufA;                                 // heads reuse a1 region
  float* phid = bufA + (long)N_NODES * 64;
  float* vout = bufA + (long)N_NODES * 128;           // [65536,3]
  float* pout = vout + (long)N_NODES * 3;             // [65536,1]
  float* dout = (float*)d_out;

  auto gemm = [&](const float* A, int lda, const float* W, float* Cp, int ldc,
                  int K, int N, const float* bias,
                  const float* g, const float* b, const float* m, const float* v,
                  int relu, int accum) {
    dim3 grid(N_NODES / 64, (N + 63) / 64);
    k_gemm<<<grid, dim3(256), 0, stream>>>(A, lda, W, Cp, ldc, K, N,
                                           bias, g, b, m, v, relu, accum);
  };

  // --- GIN conv stack: h written directly into xs columns (ld=512) ---
  const float* hin = X; int ldh = 16; int Cc = 16;
  for (int l = 0; l < 4; ++l) {
    int totC = N_NODES * (Cc >> 2);
    k_copy<<<(totC + 255) / 256, 256, 0, stream>>>(hin, ldh, z, Cc, N_NODES);
    long tote = (long)NEDGES * (Cc >> 2);
    k_edge<<<(int)((tote + 255) / 256), 256, 0, stream>>>(esrc, edst, hin, ldh, z, Cc);
    // t = relu(bn1(z @ w1 + b1))
    gemm(z, Cc, cw1[l], t, 128, Cc, 128, cb1[l], cg[l], cbb[l], cm[l], cv[l], 1, 0);
    // xs[:,l*128:] = relu(norm_l(t @ w2 + b2))   (conv -> norm -> act fused)
    gemm(t, 128, cw2[l], xs + l * 128, 512, 128, 128, cb2[l],
         ng[l], nb[l], nm[l], nv[l], 1, 0);
    hin = xs + l * 128; ldh = 512; Cc = 128;
  }

  // --- jk linear: jk = xs @ jk_w + jk_b ---
  gemm(xs, 512, jkw, jk, 512, 512, 512, jkb, nullptr, nullptr, nullptr, nullptr, 0, 0);

  // --- l1 over virtual concat [x | jk] (528 cols) split into two K-passes ---
  gemm(X, 16, l1w, a1, 256, 16, 256, nullptr, nullptr, nullptr, nullptr, nullptr, 0, 0);
  gemm(jk, 512, l1w + 16 * 256, a1, 256, 512, 256, l1b,
       nullptr, nullptr, nullptr, nullptr, 1, 1);   // accumulate, +bias, ReLU

  // --- GraphNorm (stats over batch dim per (node, channel)) ---
  k_graphnorm<<<(NODES_PER * 256 + 255) / 256, 256, 0, stream>>>(a1, gng, gnb, gna, 256);

  // --- l2: h2 = a1 @ l2w + l2b ---
  gemm(a1, 256, l2w, bufB, 128, 256, 128, l2b, nullptr, nullptr, nullptr, nullptr, 0, 0);

  // --- heads (plain linears, no activation) ---
  gemm(bufB, 128, v1w, vhid, 64, 128, 64, v1b, nullptr, nullptr, nullptr, nullptr, 0, 0);
  gemm(vhid, 64, v2w, vout, 3, 64, 3, v2b, nullptr, nullptr, nullptr, nullptr, 0, 0);
  gemm(bufB, 128, p1w, phid, 64, 128, 64, p1b, nullptr, nullptr, nullptr, nullptr, 0, 0);
  gemm(phid, 64, p2w, pout, 1, 64, 1, p2b, nullptr, nullptr, nullptr, nullptr, 0, 0);

  // --- pooling + log_softmax outputs ---
  k_phead<<<BATCH, 256, 0, stream>>>(pout, dout);                 // [64,1024]
  k_vhead<<<BATCH, 128, 0, stream>>>(vout, dout + N_NODES);       // [64,3]
}
